// SchnetInteractionBlock_21354577395850
// MI455X (gfx1250) — matrix operand order repaired and checked by reading, not compiled, not run on hardware
//
#include <hip/hip_runtime.h>
#include <hip/hip_bf16.h>

#define HIDDEN     128
#define NBASIS     50
#define NBASIS_PAD 64
#define NF         128
#define NNODES     20000
#define NEDGES     640000
#define LOG2F_     0.6931471805599453f
#define PI_OVER_CUTOFF 0.31415926535897932f

typedef __attribute__((ext_vector_type(16))) __bf16 v16bf;
typedef __attribute__((ext_vector_type(8)))  float  v8f;

__device__ __forceinline__ float sspf(float x) {
  // shifted softplus, stable: max(x,0) + log(1 + exp(-|x|)) - log(2).
  // arg of log is in [1,2] so hardware v_log_f32/v_exp_f32 are accurate enough
  // (the GEMMs already run in bf16).
  return fmaxf(x, 0.f) + __logf(1.f + __expf(-fabsf(x))) - LOG2F_;
}

// Packed B-fragment index for V_WMMA_F32_16X16X32_BF16.
// B is logically [K x 128] row-major; fragments are stored so that lane L of
// the wave working on column block cb, k-step ks loads its 16 bf16 elements
// contiguously (32 bytes) at Bf[(cb*ksteps + ks)*32 + L].
__device__ __forceinline__ int frag_idx(int k, int col, int ksteps) {
  const int colBlock = col >> 4;
  const int row      = col & 15;
  const int kstep    = k >> 5;
  const int krem     = k & 31;
  const int lane     = row + ((krem & 8) << 1);   // lanes 0-15: K 0-7/16-23, 16-31: K 8-15/24-31
  const int i        = (krem & 7) | ((krem & 16) >> 1);
  return ((colBlock * ksteps + kstep) * 32 + lane) * 16 + i;
}

// One 16x16 output tile per wave: D += A(16xK, LDS bf16, row-major, ld=lda)
//                                    * B (pre-packed fragments)
template<int K>
__device__ __forceinline__ v8f wmma_tile(const __bf16* sA, int lda,
                                         const v16bf* __restrict__ Bf,
                                         int colBlock, v8f acc, int lane)
{
  const int  row = lane & 15;
  const bool hi  = lane >= 16;
  #pragma unroll
  for (int ks = 0; ks < K / 32; ++ks) {
    v16bf a;
    // A 16x32 bf16 layout: lane half selects K 0-7/16-23 vs 8-15/24-31
    const int ka = ks * 32 + (hi ? 8 : 0);
    #pragma unroll
    for (int i = 0; i < 8; ++i) {
      a[i]     = sA[row * lda + ka + i];
      a[i + 8] = sA[row * lda + ka + 16 + i];
    }
    const v16bf b = Bf[(colBlock * (K / 32) + ks) * 32 + lane];
    acc = __builtin_amdgcn_wmma_f32_16x16x32_bf16(
        /*neg_a=*/false, a, /*neg_b=*/false, b,
        /*c_mod=*/(short)0, acc, /*reuse_a=*/false, /*reuse_b=*/false);
  }
  return acc;
}

// ---- prep: fp32 weights -> bf16 pre-packed WMMA B-fragments ----------------
__global__ void prep_kernel(const float* __restrict__ w1, const float* __restrict__ w2,
                            const float* __restrict__ l1, const float* __restrict__ l2,
                            const float* __restrict__ lw,
                            __bf16* w1p, __bf16* w2b, __bf16* l1b, __bf16* l2b, __bf16* lwb)
{
  int i = blockIdx.x * 256 + threadIdx.x;              // 0 .. 16383
  if (i < NBASIS_PAD * NF) {                           // K padded 50 -> 64
    int k = i >> 7, c = i & 127;
    float v = (k < NBASIS) ? w1[k * NF + c] : 0.f;
    w1p[frag_idx(k, c, NBASIS_PAD / 32)] = (__bf16)v;
  }
  if (i < NF * NF) {
    int k = i >> 7, c = i & 127;
    int fi = frag_idx(k, c, NF / 32);
    w2b[fi] = (__bf16)w2[i];
    l1b[fi] = (__bf16)l1[i];
    l2b[fi] = (__bf16)l2[i];
    lwb[fi] = (__bf16)lw[i];
  }
}

// ---- zero the scatter accumulator ------------------------------------------
__global__ void zero_kernel(float* agg) {
  int i = blockIdx.x * 256 + threadIdx.x;
  if (i < NNODES * NF) agg[i] = 0.f;
}

// ---- h = x @ lin1_w  (20000x128 @ 128x128) ---------------------------------
__global__ void __launch_bounds__(256) lin1_kernel(const float* __restrict__ x,
                                                   const __bf16* __restrict__ l1b,
                                                   float* __restrict__ h)
{
  __shared__ __bf16 sA[16 * 128];
  const int r0 = blockIdx.x * 16;
  {
    // 2048 elems = 256 threads x 8 consecutive -> b128 loads, ds_store_b128
    const float* sp = x + (size_t)r0 * HIDDEN + threadIdx.x * 8;
    __bf16* dp = sA + threadIdx.x * 8;
    #pragma unroll
    for (int j = 0; j < 8; ++j) dp[j] = (__bf16)sp[j];
  }
  __syncthreads();

  const int lane = threadIdx.x & 31, w = threadIdx.x >> 5;
  const int row = lane & 15;
  const bool hi = lane >= 16;

  v8f acc = {};
  acc = wmma_tile<128>(sA, 128, (const v16bf*)l1b, w, acc, lane);

  #pragma unroll
  for (int v = 0; v < 8; ++v)
    h[(r0 + v + (hi ? 8 : 0)) * NF + w * 16 + row] = acc[v];
}

// ---- fused edge pipeline ---------------------------------------------------
__global__ void __launch_bounds__(256) edge_kernel(const float* __restrict__ ea,
                                                   const float* __restrict__ ew,
                                                   const int* __restrict__ src,
                                                   const int* __restrict__ dst,
                                                   const __bf16* __restrict__ w1p,
                                                   const float* __restrict__ b1,
                                                   const __bf16* __restrict__ w2b,
                                                   const float* __restrict__ b2,
                                                   const float* __restrict__ h,
                                                   float* __restrict__ agg)
{
  __shared__ __bf16 sA[16 * NBASIS_PAD];   // edge_attr tile, K padded to 64
  __shared__ __bf16 sH[16 * NF];           // intermediate ssp(...@w1+b1) tile

  const int e0 = blockIdx.x * 16;
  {
    // 1024 elems = 256 threads x 4 consecutive
    const int base = threadIdx.x * 4;
    const int r = base >> 6, k = base & 63;
    __bf16* dp = sA + base;
    #pragma unroll
    for (int j = 0; j < 4; ++j) {
      float v = (k + j < NBASIS) ? ea[(e0 + r) * NBASIS + k + j] : 0.f;
      dp[j] = (__bf16)v;
    }
  }
  __syncthreads();

  const int lane = threadIdx.x & 31, w = threadIdx.x >> 5;
  const int row = lane & 15;
  const bool hi = lane >= 16;

  // GEMM1: [16,64] x [64,128]
  v8f acc = {};
  acc = wmma_tile<NBASIS_PAD>(sA, NBASIS_PAD, (const v16bf*)w1p, w, acc, lane);

  const float bias1 = b1[w * 16 + row];
  #pragma unroll
  for (int v = 0; v < 8; ++v) {
    float xv = sspf(acc[v] + bias1);
    sH[(v + (hi ? 8 : 0)) * NF + w * 16 + row] = (__bf16)xv;
  }
  __syncthreads();

  // GEMM2: [16,128] x [128,128]
  v8f acc2 = {};
  acc2 = wmma_tile<NF>(sH, NF, (const v16bf*)w2b, w, acc2, lane);

  const float bias2 = b2[w * 16 + row];
  const int filt = w * 16 + row;
  #pragma unroll
  for (int v = 0; v < 8; ++v) {
    const int e = e0 + v + (hi ? 8 : 0);
    const float C  = 0.5f * (__cosf(ew[e] * PI_OVER_CUTOFF) + 1.f);
    const float Wv = (acc2[v] + bias2) * C;
    const float msg = h[src[e] * NF + filt] * Wv;
    atomicAdd(&agg[dst[e] * NF + filt], msg);
  }
}

// ---- tail: out = ssp(agg@lin2+b2) @ lin + b --------------------------------
__global__ void __launch_bounds__(256) tail_kernel(const float* __restrict__ agg,
                                                   const __bf16* __restrict__ l2b,
                                                   const float* __restrict__ lin2_b,
                                                   const __bf16* __restrict__ lwb,
                                                   const float* __restrict__ lin_b,
                                                   float* __restrict__ out)
{
  __shared__ __bf16 sA[16 * NF];
  __shared__ __bf16 sH[16 * HIDDEN];

  const int r0 = blockIdx.x * 16;
  {
    const float* sp = agg + (size_t)r0 * NF + threadIdx.x * 8;
    __bf16* dp = sA + threadIdx.x * 8;
    #pragma unroll
    for (int j = 0; j < 8; ++j) dp[j] = (__bf16)sp[j];
  }
  __syncthreads();

  const int lane = threadIdx.x & 31, w = threadIdx.x >> 5;
  const int row = lane & 15;
  const bool hi = lane >= 16;

  v8f acc = {};
  acc = wmma_tile<NF>(sA, NF, (const v16bf*)l2b, w, acc, lane);

  const float bias1 = lin2_b[w * 16 + row];
  #pragma unroll
  for (int v = 0; v < 8; ++v) {
    float xv = sspf(acc[v] + bias1);
    sH[(v + (hi ? 8 : 0)) * HIDDEN + w * 16 + row] = (__bf16)xv;
  }
  __syncthreads();

  v8f acc2 = {};
  acc2 = wmma_tile<HIDDEN>(sH, HIDDEN, (const v16bf*)lwb, w, acc2, lane);

  const float bias2 = lin_b[w * 16 + row];
  #pragma unroll
  for (int v = 0; v < 8; ++v)
    out[(r0 + v + (hi ? 8 : 0)) * HIDDEN + w * 16 + row] = acc2[v] + bias2;
}

// ---------------------------------------------------------------------------
extern "C" void kernel_launch(void* const* d_in, const int* in_sizes, int n_in,
                              void* d_out, int out_size, void* d_ws, size_t ws_size,
                              hipStream_t stream) {
  const float* x      = (const float*)d_in[0];
  const int*   ei     = (const int*)d_in[1];    // [2, E]: row 0 = src, row 1 = dst
  const float* ew     = (const float*)d_in[2];
  const float* ea     = (const float*)d_in[3];
  const float* w1     = (const float*)d_in[4];
  const float* b1     = (const float*)d_in[5];
  const float* w2     = (const float*)d_in[6];
  const float* b2     = (const float*)d_in[7];
  const float* l1     = (const float*)d_in[8];
  const float* l2     = (const float*)d_in[9];
  const float* l2bias = (const float*)d_in[10];
  const float* lw     = (const float*)d_in[11];
  const float* lwbias = (const float*)d_in[12];
  float* out = (float*)d_out;

  const int* src = ei;
  const int* dst = ei + NEDGES;

  // workspace layout (~20.6 MB); all offsets 32-byte aligned for v16bf loads
  char* ws = (char*)d_ws;
  float*  h    = (float*)ws;  ws += (size_t)NNODES * NF * sizeof(float);
  float*  agg  = (float*)ws;  ws += (size_t)NNODES * NF * sizeof(float);
  __bf16* w1p  = (__bf16*)ws; ws += (size_t)NBASIS_PAD * NF * sizeof(__bf16);
  __bf16* w2b  = (__bf16*)ws; ws += (size_t)NF * NF * sizeof(__bf16);
  __bf16* l1b  = (__bf16*)ws; ws += (size_t)HIDDEN * NF * sizeof(__bf16);
  __bf16* l2b  = (__bf16*)ws; ws += (size_t)NF * HIDDEN * sizeof(__bf16);
  __bf16* lwb  = (__bf16*)ws;

  prep_kernel<<<64, 256, 0, stream>>>(w1, w2, l1, l2, lw, w1p, w2b, l1b, l2b, lwb);
  zero_kernel<<<(NNODES * NF + 255) / 256, 256, 0, stream>>>(agg);
  lin1_kernel<<<NNODES / 16, 256, 0, stream>>>(x, l1b, h);
  edge_kernel<<<NEDGES / 16, 256, 0, stream>>>(ea, ew, src, dst, w1p, b1, w2b, b2, h, agg);
  tail_kernel<<<NNODES / 16, 256, 0, stream>>>(agg, l2b, l2bias, lwb, lwbias, out);
}
